// PclCldLoss_77695958384730
// MI455X (gfx1250) — matrix operand order repaired and checked by reading, not compiled, not run on hardware
//
#include <hip/hip_runtime.h>
#include <hip/hip_bf16.h>

typedef __attribute__((ext_vector_type(16))) _Float16 v16h;
typedef __attribute__((ext_vector_type(8)))  _Float16 v8h;
typedef __attribute__((ext_vector_type(8)))  float    v8f;

#define BSZ   1024
#define DDIM  64
#define MNUM  3
#define CNUM  16384
#define WAVES 8
#define LOG2E 1.4426950408889634f

// Workspace layout
constexpr size_t SUMEXP_ELEMS = size_t(2) * MNUM * BSZ;            // 6144 f32
constexpr size_t SUMEXP_BYTES = SUMEXP_ELEMS * 4;                  // 24 KB
constexpr size_t RCP_ELEMS    = size_t(2) * MNUM * CNUM;           // 98304 f32
constexpr size_t RCP_BYTES    = RCP_ELEMS * 4;                     // 384 KB
constexpr size_t F16_OFF      = SUMEXP_BYTES + RCP_BYTES;          // 417792 (256B aligned)
constexpr size_t F16_ELEMS    = size_t(2) * MNUM * CNUM * DDIM;    // 6.29M halves
constexpr size_t F16_BYTES    = F16_ELEMS * 2;                     // 12.58 MB

// ---------------------------------------------------------------------------
// Fragment loaders per CDNA5 ISA VGPR layouts.
// A 16x32 f16: lanes 0-15: row=lane, K {0..7},{16..23}; lanes 16-31: K {8..15},{24..31}.
// Caller passes p = rowBase + koff, koff = (lane<16)?0:8.
__device__ __forceinline__ v16h load_fragA_f16(const _Float16* p) {
  v8h a = *reinterpret_cast<const v8h*>(p);
  v8h b = *reinterpret_cast<const v8h*>(p + 16);
  v16h h;
  #pragma unroll
  for (int i = 0; i < 8; ++i) { h[i] = a[i]; h[8 + i] = b[i]; }
  return h;
}

__device__ __forceinline__ v16h load_fragA_f32(const float* p) {
  float4 x0 = *reinterpret_cast<const float4*>(p);
  float4 x1 = *reinterpret_cast<const float4*>(p + 4);
  float4 x2 = *reinterpret_cast<const float4*>(p + 16);
  float4 x3 = *reinterpret_cast<const float4*>(p + 20);
  v16h h;
  h[0]  = (_Float16)x0.x; h[1]  = (_Float16)x0.y; h[2]  = (_Float16)x0.z; h[3]  = (_Float16)x0.w;
  h[4]  = (_Float16)x1.x; h[5]  = (_Float16)x1.y; h[6]  = (_Float16)x1.z; h[7]  = (_Float16)x1.w;
  h[8]  = (_Float16)x2.x; h[9]  = (_Float16)x2.y; h[10] = (_Float16)x2.z; h[11] = (_Float16)x2.w;
  h[12] = (_Float16)x3.x; h[13] = (_Float16)x3.y; h[14] = (_Float16)x3.z; h[15] = (_Float16)x3.w;
  return h;
}

// B 32x16 f16: lane = column N (mod 16); lanes 0-15 K {0..7},{16..23}; lanes 16-31 K {8..15},{24..31}.
// Caller passes p = colBase + ((lane<16)?0:8).
__device__ __forceinline__ v16h load_fragB_f32(const float* p) {
  float4 x0 = *reinterpret_cast<const float4*>(p);
  float4 x1 = *reinterpret_cast<const float4*>(p + 4);
  float4 x2 = *reinterpret_cast<const float4*>(p + 16);
  float4 x3 = *reinterpret_cast<const float4*>(p + 20);
  v16h h;
  h[0]  = (_Float16)x0.x; h[1]  = (_Float16)x0.y; h[2]  = (_Float16)x0.z; h[3]  = (_Float16)x0.w;
  h[4]  = (_Float16)x1.x; h[5]  = (_Float16)x1.y; h[6]  = (_Float16)x1.z; h[7]  = (_Float16)x1.w;
  h[8]  = (_Float16)x2.x; h[9]  = (_Float16)x2.y; h[10] = (_Float16)x2.z; h[11] = (_Float16)x2.w;
  h[12] = (_Float16)x3.x; h[13] = (_Float16)x3.y; h[14] = (_Float16)x3.z; h[15] = (_Float16)x3.w;
  return h;
}

#define WMMA_F16(A, B, C) \
  __builtin_amdgcn_wmma_f32_16x16x32_f16(false, (A), false, (B), (short)0, (C), false, false)

// ---------------------------------------------------------------------------
__global__ __launch_bounds__(256) void pcl_init(float* __restrict__ sumExp,
                                                float* __restrict__ out) {
  int i = blockIdx.x * blockDim.x + threadIdx.x;
  if (i < (int)SUMEXP_ELEMS) sumExp[i] = 0.0f;
  if (i == 0) out[0] = 0.0f;
}

// rcpc[i] = log2(e) / conc[i]  (both branches concatenated)
__global__ __launch_bounds__(256) void pcl_prep_rcp(const float* __restrict__ a,
                                                    const float* __restrict__ b,
                                                    float* __restrict__ dst) {
  int i = blockIdx.x * blockDim.x + threadIdx.x;
  const int N = MNUM * CNUM;
  if (i < 2 * N) {
    float v = (i < N) ? a[i] : b[i - N];
    dst[i] = LOG2E / v;
  }
}

// f16 copy of both centroid tensors, concatenated [branch][m][c][d]
__global__ __launch_bounds__(256) void pcl_prep_f16(const float* __restrict__ a,
                                                    const float* __restrict__ b,
                                                    _Float16* __restrict__ dst) {
  size_t i = (size_t)blockIdx.x * blockDim.x + threadIdx.x;   // quad index
  const size_t NQ = (size_t)MNUM * CNUM * DDIM / 4;
  if (i < 2 * NQ) {
    const float* src = (i < NQ) ? a : b;
    size_t j = (i < NQ) ? i : i - NQ;
    float4 v = reinterpret_cast<const float4*>(src)[j];
    _Float16* o = dst + i * 4;
    o[0] = (_Float16)v.x; o[1] = (_Float16)v.y;
    o[2] = (_Float16)v.z; o[3] = (_Float16)v.w;
  }
}

// ---------------------------------------------------------------------------
// Cached main kernel: f16 centroids + precomputed log2e/conc.
// Grid: branch(2) * m(3) * bpair(32) * csplit(2) = 384 blocks, 256 threads.
// Each wave owns two 16-wide b-tiles (fragments register-resident) and sweeps
// 64 c-tiles; per tile: 4 b128 A-loads, 2 b128 rcp-loads, 4 WMMAs, 16 exp2.
__global__ __launch_bounds__(256) void pcl_gemm_exp_cached(
    const _Float16* __restrict__ cen16,       // [2][M][C][D] halves
    const float* __restrict__ feat0,          // branch-0 features (features_I)
    const float* __restrict__ feat1,          // branch-1 features (features)
    const float* __restrict__ rcpc,           // [2][M][C] log2e/conc
    float* __restrict__ sumExp) {             // [2][M][B]
  const int lane = threadIdx.x & 31;
  const int wave = threadIdx.x >> 5;
  const int bp   = blockIdx.x & 31;                 // 32 b-pairs
  const int cs   = (blockIdx.x >> 5) & 1;           // C split
  const int mm   = (blockIdx.x >> 6) % MNUM;
  const int br   = blockIdx.x / (32 * 2 * MNUM);

  const float* feat = br ? feat1 : feat0;
  const int b0  = bp << 5;                          // 32 columns: tiles b0, b0+16
  const int col = lane & 15;

  // Feature fragments for both b-tiles, both K slices (register resident).
  const int kbaseB = (lane < 16) ? 0 : 8;
  const float* f0p = feat + (size_t)(b0 + col) * DDIM + kbaseB;
  const float* f1p = feat + (size_t)(b0 + 16 + col) * DDIM + kbaseB;
  const v16h fb00 = load_fragB_f32(f0p);
  const v16h fb01 = load_fragB_f32(f0p + 32);
  const v16h fb10 = load_fragB_f32(f1p);
  const v16h fb11 = load_fragB_f32(f1p + 32);

  const int koffA  = (lane < 16) ? 0 : 8;
  const int rowOff = (lane < 16) ? 0 : 8;
  const _Float16* cbase = cen16 + ((size_t)br * MNUM + mm) * CNUM * DDIM;
  const float*    rbase = rcpc  + ((size_t)br * MNUM + mm) * CNUM;

  float s0a = 0.0f, s0b = 0.0f, s1a = 0.0f, s1b = 0.0f;
  const int cstep = 16 * WAVES * 2;                 // 256
  for (int c0 = (cs * WAVES + wave) * 16; c0 < CNUM; c0 += cstep) {
    const _Float16* ap = cbase + (size_t)(c0 + col) * DDIM + koffA;
    __builtin_prefetch(ap + (size_t)cstep * DDIM, 0, 1);
    const v16h fa0 = load_fragA_f16(ap);
    const v16h fa1 = load_fragA_f16(ap + 32);

    v8f acc0 = {}, acc1 = {};
    acc0 = WMMA_F16(fa0, fb00, acc0);
    acc0 = WMMA_F16(fa1, fb01, acc0);
    acc1 = WMMA_F16(fa0, fb10, acc1);
    acc1 = WMMA_F16(fa1, fb11, acc1);

    const float* rp = rbase + c0 + rowOff;
    float4 r0 = *reinterpret_cast<const float4*>(rp);
    float4 r1 = *reinterpret_cast<const float4*>(rp + 4);
    float rc[8] = {r0.x, r0.y, r0.z, r0.w, r1.x, r1.y, r1.z, r1.w};
    #pragma unroll
    for (int i = 0; i < 8; i += 2) {
      s0a += __builtin_amdgcn_exp2f(acc0[i]     * rc[i]);
      s0b += __builtin_amdgcn_exp2f(acc0[i + 1] * rc[i + 1]);
      s1a += __builtin_amdgcn_exp2f(acc1[i]     * rc[i]);
      s1b += __builtin_amdgcn_exp2f(acc1[i + 1] * rc[i + 1]);
    }
  }

  float s0 = s0a + s0b, s1 = s1a + s1b;
  s0 += __shfl_xor(s0, 16, 32);
  s1 += __shfl_xor(s1, 16, 32);
  if (lane < 16) {
    float* dst = &sumExp[((size_t)br * MNUM + mm) * BSZ + b0 + lane];
    atomicAdd(dst, s0);
    atomicAdd(dst + 16, s1);
  }
}

// ---------------------------------------------------------------------------
// Fallback (small workspace): inline f32->f16 conversion, inline rcp.
__global__ __launch_bounds__(256) void pcl_gemm_exp_f32(
    const float* __restrict__ cen0, const float* __restrict__ feat0, const float* __restrict__ conc0,
    const float* __restrict__ cen1, const float* __restrict__ feat1, const float* __restrict__ conc1,
    float* __restrict__ sumExp) {
  const int lane  = threadIdx.x & 31;
  const int wave  = threadIdx.x >> 5;
  const int btile = blockIdx.x & 63;
  const int mm    = (blockIdx.x >> 6) % MNUM;
  const int br    = blockIdx.x / (64 * MNUM);
  const float* cen  = br ? cen1  : cen0;
  const float* feat = br ? feat1 : feat0;
  const float* conc = br ? conc1 : conc0;

  const int b0  = btile << 4;
  const int col = lane & 15;
  const int kbaseB = (lane < 16) ? 0 : 8;
  const float* fptr = feat + (size_t)(b0 + col) * DDIM + kbaseB;
  const v16h fb0 = load_fragB_f32(fptr);
  const v16h fb1 = load_fragB_f32(fptr + 32);

  const int koffA  = (lane < 16) ? 0 : 8;
  const int rowOff = (lane < 16) ? 0 : 8;
  const float* cbase = cen + (size_t)mm * CNUM * DDIM;
  const float* concm = conc + (size_t)mm * CNUM;

  float accSum = 0.0f;
  for (int c0 = wave * 16; c0 < CNUM; c0 += 16 * WAVES) {
    const float* ap = cbase + (size_t)(c0 + col) * DDIM + koffA;
    __builtin_prefetch(ap + 16 * WAVES * DDIM, 0, 1);
    const v16h fa0 = load_fragA_f32(ap);
    const v16h fa1 = load_fragA_f32(ap + 32);
    v8f acc = {};
    acc = WMMA_F16(fa0, fb0, acc);
    acc = WMMA_F16(fa1, fb1, acc);
    const float* cp = concm + c0 + rowOff;
    #pragma unroll
    for (int i = 0; i < 8; ++i) {
      float rc = __builtin_amdgcn_rcpf(cp[i]) * LOG2E;
      accSum += __builtin_amdgcn_exp2f(acc[i] * rc);
    }
  }
  accSum += __shfl_xor(accSum, 16, 32);
  if (lane < 16)
    atomicAdd(&sumExp[((size_t)br * MNUM + mm) * BSZ + b0 + lane], accSum);
}

// ---------------------------------------------------------------------------
// Positive-logit gather + final reduction (fp32 exact dot for the positives).
__global__ __launch_bounds__(256) void pcl_pos_loss(
    const float* __restrict__ cen0, const float* __restrict__ feat0,
    const float* __restrict__ conc0, const int* __restrict__ lab0,
    const float* __restrict__ cen1, const float* __restrict__ feat1,
    const float* __restrict__ conc1, const int* __restrict__ lab1,
    const float* __restrict__ sumExp, const int* __restrict__ lb,
    float* __restrict__ out) {
  int idx = blockIdx.x * blockDim.x + threadIdx.x;
  float contrib = 0.0f;
  if (idx < 2 * MNUM * BSZ) {
    int br  = idx / (MNUM * BSZ);
    int rem = idx - br * MNUM * BSZ;
    int mm  = rem / BSZ;
    int b   = rem - mm * BSZ;
    const float* cen  = br ? cen1  : cen0;
    const float* feat = br ? feat1 : feat0;
    const float* conc = br ? conc1 : conc0;
    const int*   lab  = br ? lab1  : lab0;

    int c = lab[(size_t)mm * BSZ + b];
    const float4* cp = reinterpret_cast<const float4*>(cen + ((size_t)mm * CNUM + c) * DDIM);
    const float4* fp = reinterpret_cast<const float4*>(feat + (size_t)b * DDIM);
    float dot = 0.0f;
    #pragma unroll
    for (int i = 0; i < DDIM / 4; ++i) {
      float4 x = cp[i], y = fp[i];
      dot += x.x * y.x + x.y * y.y + x.z * y.z + x.w * y.w;
    }
    float logit = dot / conc[(size_t)mm * CNUM + c];
    contrib = logit - __logf(sumExp[idx]);
  }
  #pragma unroll
  for (int off = 16; off > 0; off >>= 1) contrib += __shfl_down(contrib, off, 32);
  if ((threadIdx.x & 31) == 0) {
    float scale = -0.5f * (float)lb[0] / ((float)BSZ * (float)MNUM);
    atomicAdd(out, contrib * scale);
  }
}

// ---------------------------------------------------------------------------
extern "C" void kernel_launch(void* const* d_in, const int* in_sizes, int n_in,
                              void* d_out, int out_size, void* d_ws, size_t ws_size,
                              hipStream_t stream) {
  const float* feat   = (const float*)d_in[0];
  const float* feat_I = (const float*)d_in[1];
  const float* cen    = (const float*)d_in[2];
  const float* cen_I  = (const float*)d_in[3];
  const float* conc   = (const float*)d_in[4];
  const float* conc_I = (const float*)d_in[5];
  const int*   lab    = (const int*)d_in[6];
  const int*   lab_I  = (const int*)d_in[7];
  const int*   lb     = (const int*)d_in[8];
  float* out    = (float*)d_out;
  float* sumExp = (float*)d_ws;

  pcl_init<<<(int)((SUMEXP_ELEMS + 255) / 256), 256, 0, stream>>>(sumExp, out);

  const bool cached = (ws_size >= F16_OFF + F16_BYTES);
  if (cached) {
    float*     rcpc  = (float*)((char*)d_ws + SUMEXP_BYTES);
    _Float16*  cen16 = (_Float16*)((char*)d_ws + F16_OFF);
    pcl_prep_rcp<<<(int)((RCP_ELEMS + 255) / 256), 256, 0, stream>>>(conc, conc_I, rcpc);
    pcl_prep_f16<<<(int)((F16_ELEMS / 4 + 255) / 256), 256, 0, stream>>>(cen, cen_I, cen16);
    // Branch 0: (M_kmeans, features_I); Branch 1: (M_kmeans_I, features)
    pcl_gemm_exp_cached<<<2 * MNUM * 32 * 2, 256, 0, stream>>>(
        cen16, feat_I, feat, rcpc, sumExp);
  } else {
    pcl_gemm_exp_f32<<<2 * MNUM * (BSZ / 16), 256, 0, stream>>>(
        cen, feat_I, conc, cen_I, feat, conc_I, sumExp);
  }

  pcl_pos_loss<<<(2 * MNUM * BSZ + 255) / 256, 256, 0, stream>>>(
      cen, feat_I, conc, lab, cen_I, feat, conc_I, lab_I, sumExp, lb, out);
}
